// T5Attention_33483565040068
// MI455X (gfx1250) — matrix operand order repaired and checked by reading, not compiled, not run on hardware
//
#include <hip/hip_runtime.h>
#include <hip/hip_bf16.h>
#include <stdint.h>

// ---------------------------------------------------------------------------
// T5 attention for MI455X (gfx1250), wave32, WMMA bf16 path + TDM bias tiles.
//   B=2, L=2048, DIM=1024, NH=16, HD=64
// Pipeline:
//   1) cvt x -> bf16 ; transpose wq/wk/wv/wo -> bf16 W^T[n][k]
//   2) WMMA GEMM: Q,K as [B,NH,L,64] bf16 ; V transposed [B,NH,64,L] bf16
//   3) fused flash attention; pos_bias tiles staged LDS via tensor_load_to_lds
//   4) WMMA GEMM: attn_out(bf16) @ wo -> f32 d_out
// ---------------------------------------------------------------------------

#define B_   2
#define L_   2048
#define DIM_ 1024
#define NH_  16
#define HD_  64
#define M_   (B_ * L_)
#define NEGINF_ (-3.4028234663852886e38f)

typedef __bf16 bf16;
typedef uint32_t u32;
typedef __attribute__((ext_vector_type(16))) __bf16 v16bf;
typedef __attribute__((ext_vector_type(8)))  __bf16 v8bf;
typedef __attribute__((ext_vector_type(8)))  float  v8f;
typedef __attribute__((ext_vector_type(4)))  u32    u32x4;
typedef __attribute__((ext_vector_type(8)))  int    i32x8;
typedef __attribute__((ext_vector_type(4)))  int    i32x4;

// ---- WMMA wrapper: D = A(16x32 bf16) * B(32x16 bf16) + C(16x16 f32) -------
__device__ __forceinline__ v8f wmma_bf16(v16bf a, v16bf b, v8f c) {
  return __builtin_amdgcn_wmma_f32_16x16x32_bf16(
      false, a, false, b, (short)0, c, false, false);
}

// ---- A-fragment (16x32, M x K) from row-major [.. x lda] -------------------
__device__ __forceinline__ v16bf load_a_frag(const bf16* base, int lda,
                                             int row0, int k0, int lane) {
  const int lr = lane & 15, hf = lane >> 4;
  const bf16* p = base + (size_t)(row0 + lr) * lda + (k0 + hf * 8);
  v8bf lo = *(const v8bf*)p;
  v8bf hi = *(const v8bf*)(p + 16);
  v16bf a;
#pragma unroll
  for (int i = 0; i < 8; ++i) { a[i] = lo[i]; a[i + 8] = hi[i]; }
  return a;
}

// ---- B-fragment (32x16, K x N) from transposed storage BT[n][k] ------------
__device__ __forceinline__ v16bf load_bt_frag(const bf16* base, size_t ldb,
                                              int col0, int k0, int lane) {
  const int lr = lane & 15, hf = lane >> 4;
  const bf16* p = base + (size_t)(col0 + lr) * ldb + (k0 + hf * 8);
  v8bf lo = *(const v8bf*)p;
  v8bf hi = *(const v8bf*)(p + 16);
  v16bf b;
#pragma unroll
  for (int i = 0; i < 8; ++i) { b[i] = lo[i]; b[i + 8] = hi[i]; }
  return b;
}

// ---- TDM: DMA a 16-row x 32-col f32 tile from a [2048 x 2048] tensor ------
// D# built per CDNA5 ISA ch.8: group0 {count, lds_addr, global_addr, type=2},
// group1 {data_size=4B, tensor_dim0/1=2048, tile 32x16, dim0_stride=2048}.
__device__ __forceinline__ void tdm_load_bias_tile(const float* gptr, u32 lds_off) {
  const u32 lo = __builtin_amdgcn_readfirstlane((u32)(uintptr_t)gptr);
  const u32 hi = __builtin_amdgcn_readfirstlane((u32)(((uintptr_t)gptr) >> 32));
  const u32 ld = __builtin_amdgcn_readfirstlane(lds_off);

  u32x4 g0;
  g0[0] = 1u;                                   // count=1, user descriptor
  g0[1] = ld;                                   // lds_addr (bytes)
  g0[2] = lo;                                   // global_addr[31:0]
  g0[3] = (hi & 0x01FFFFFFu) | (2u << 30);      // global_addr[56:32] | type=2

  i32x8 g1;
  g1[0] = (int)(2u << 16);                      // data_size = 2 (4 bytes)
  g1[1] = (int)((2048u & 0xFFFFu) << 16);       // tensor_dim0[15:0] in [31:16]
  g1[2] = (int)((2048u >> 16) | ((2048u & 0xFFFFu) << 16)); // dim0 hi | dim1 lo
  g1[3] = (int)((2048u >> 16) | (32u << 16));   // dim1 hi | tile_dim0 = 32
  g1[4] = 16;                                   // tile_dim1 = 16, tile_dim2 = 0
  g1[5] = 2048;                                 // tensor_dim0_stride[31:0]
  g1[6] = 0;                                    // stride hi | dim1_stride lo
  g1[7] = 0;

  i32x4 z4 = (i32x4)0;
#if __clang_major__ >= 23
  i32x8 z8 = (i32x8)0;
  __builtin_amdgcn_tensor_load_to_lds(g0, g1, z4, z4, z8, 0);
#else
  __builtin_amdgcn_tensor_load_to_lds(g0, g1, z4, z4, 0);
#endif
}

// ---------------------------------------------------------------------------
__global__ __launch_bounds__(256) void cvt_f32_bf16(const float* __restrict__ in,
                                                    bf16* __restrict__ out, int n) {
  int i = blockIdx.x * 256 + threadIdx.x;
  if (i < n) out[i] = (bf16)in[i];
}

__global__ __launch_bounds__(256) void transpose_w_bf16(const float* __restrict__ w,
                                                        bf16* __restrict__ wT) {
  int n = blockIdx.x;
  for (int k = threadIdx.x; k < DIM_; k += 256)
    wT[(size_t)n * DIM_ + k] = (bf16)w[(size_t)k * DIM_ + n];
}

// ---------------------------------------------------------------------------
// GEMM: C[M x N] = A[M x K] * BT^T ; one wave per 16x16 tile, 8 waves/block.
__global__ __launch_bounds__(256) void gemm_wmma_bf16(const bf16* __restrict__ A,
                                                      const bf16* __restrict__ BT,
                                                      void* __restrict__ C,
                                                      int M, int N, int K, int mode) {
  const int wid = threadIdx.x >> 5, lane = threadIdx.x & 31;
  const int tile = blockIdx.x * 8 + wid;
  const int tiles_n = N >> 4;
  const int tm = tile / tiles_n, tn = tile % tiles_n;
  if (tm >= (M >> 4)) return;
  const int row0 = tm << 4, col0 = tn << 4;

  v8f acc = {};
  for (int k0 = 0; k0 < K; k0 += 32) {
    if (k0 + 256 < K) {
      __builtin_prefetch(A + (size_t)(row0 + (lane & 15)) * K + k0 + 256, 0, 1);
      __builtin_prefetch(BT + (size_t)(col0 + (lane & 15)) * K + k0 + 256, 0, 1);
    }
    v16bf a = load_a_frag(A, K, row0, k0, lane);
    v16bf b = load_bt_frag(BT, (size_t)K, col0, k0, lane);
    acc = wmma_bf16(a, b, acc);
  }

  const int lr = lane & 15, hf = lane >> 4;
  const int col = col0 + lr;
#pragma unroll
  for (int r = 0; r < 8; ++r) {
    const int row = row0 + hf * 8 + r;
    const float v = acc[r];
    if (mode == 3) {
      ((float*)C)[(size_t)row * N + col] = v;
    } else {
      const int bb = row / L_, ii = row % L_;
      const int hh = col >> 6, dd = col & 63;
      bf16* o = (bf16*)C;
      if (mode == 2)
        o[(((size_t)bb * NH_ + hh) * HD_ + dd) * L_ + ii] = (bf16)v;
      else
        o[(((size_t)bb * NH_ + hh) * L_ + ii) * HD_ + dd] = (bf16)v;
    }
  }
}

// ---------------------------------------------------------------------------
// Fused attention. Dynamic LDS layout (byte offsets from LDS base 0):
//   [0, 16384)            : bias tiles, 4 waves x 2 buffers x (16x32 f32 = 2KB)
//   [16384, 20480)        : P tiles, 4 waves x (16x32 bf16 = 1KB)
extern __shared__ __align__(16) char smem_dyn[];

__global__ __launch_bounds__(128) void attn_wmma(const bf16* __restrict__ q,
                                                 const bf16* __restrict__ k,
                                                 const bf16* __restrict__ vT,
                                                 const float* __restrict__ pos_bias,
                                                 const int* __restrict__ mask,
                                                 bf16* __restrict__ attn_out) {
  const int wid = threadIdx.x >> 5, lane = threadIdx.x & 31;
  const int gid = blockIdx.x * 4 + wid;           // 4096 wave-tiles total
  const int bh  = gid >> 7;                       // b*NH + h
  const int i0  = (gid & 127) << 4;               // q row tile
  const int bb  = bh >> 4;

  const bf16*  qh = q  + (size_t)bh * L_ * HD_;
  const bf16*  kh = k  + (size_t)bh * L_ * HD_;
  const bf16*  vh = vT + (size_t)bh * HD_ * L_;
  const float* pb = pos_bias + (size_t)bh * L_ * L_ + (size_t)i0 * L_;
  const int*   mk = mask + bb * L_;

  const int lr = lane & 15, hf = lane >> 4;
  const u32 biasBase0 = (u32)(wid * 2) * 2048u;
  const u32 biasBase1 = biasBase0 + 2048u;
  bf16* myP = (bf16*)(smem_dyn + 16384 + wid * 1024);

  // Q fragments (16 x 64 -> two K=32 A-frags), loaded once
  const v16bf qa0 = load_a_frag(qh, HD_, i0, 0,  lane);
  const v16bf qa1 = load_a_frag(qh, HD_, i0, 32, lane);

  v8f acc0 = {}, acc1 = {}, acc2 = {}, acc3 = {};
  float m[8], l[8];
#pragma unroll
  for (int r = 0; r < 8; ++r) { m[r] = -3.0e38f; l[r] = 0.0f; }

  // prologue: DMA first bias tile into buffer 0
  tdm_load_bias_tile(pb, biasBase0);
  int buf = 0;

  for (int j0 = 0; j0 < L_; j0 += 32) {
    // pipeline: issue next tile (dummy re-issue on last iter keeps count math flat)
    const int jn = (j0 + 32 < L_) ? (j0 + 32) : j0;
    tdm_load_bias_tile(pb + jn, buf ? biasBase0 : biasBase1);

    // ---- S = Q * K^T for 32 keys ----
    v8f s0 = {}, s1 = {};
    s0 = wmma_bf16(qa0, load_bt_frag(kh, HD_, j0,      0,  lane), s0);
    s0 = wmma_bf16(qa1, load_bt_frag(kh, HD_, j0,      32, lane), s0);
    s1 = wmma_bf16(qa0, load_bt_frag(kh, HD_, j0 + 16, 0,  lane), s1);
    s1 = wmma_bf16(qa1, load_bt_frag(kh, HD_, j0 + 16, 32, lane), s1);

    // ---- current bias tile is complete once <=1 tensor op outstanding ----
    __builtin_amdgcn_s_wait_tensorcnt(1);
    const float* bias = (const float*)(smem_dyn + (buf ? biasBase1 : biasBase0));

    const int c0 = j0 + lr, c1 = j0 + 16 + lr;
    const bool ok0 = (mk[c0] != 0), ok1 = (mk[c1] != 0);
#pragma unroll
    for (int r = 0; r < 8; ++r) {
      const int rloc = hf * 8 + r;
      s0[r] += ok0 ? bias[rloc * 32 + lr]      : NEGINF_;
      s1[r] += ok1 ? bias[rloc * 32 + 16 + lr] : NEGINF_;
    }

    // ---- online softmax over this 32-key chunk ----
    float scale[8];
#pragma unroll
    for (int r = 0; r < 8; ++r) {
      float mx = fmaxf(s0[r], s1[r]);
#pragma unroll
      for (int off = 1; off < 16; off <<= 1) mx = fmaxf(mx, __shfl_xor(mx, off, 32));
      const float mnew = fmaxf(m[r], mx);
      scale[r] = __expf(m[r] - mnew);
      const float p0 = __expf(s0[r] - mnew);
      const float p1 = __expf(s1[r] - mnew);
      s0[r] = p0; s1[r] = p1;
      float sum = p0 + p1;
#pragma unroll
      for (int off = 1; off < 16; off <<= 1) sum += __shfl_xor(sum, off, 32);
      l[r] = l[r] * scale[r] + sum;
      m[r] = mnew;
    }
#pragma unroll
    for (int r = 0; r < 8; ++r) {
      acc0[r] *= scale[r]; acc1[r] *= scale[r];
      acc2[r] *= scale[r]; acc3[r] *= scale[r];
    }

    // ---- re-layout P: C-fragment -> row-major LDS -> A-fragment ----
#pragma unroll
    for (int r = 0; r < 8; ++r) {
      const int row = hf * 8 + r;
      myP[row * 32 + lr]      = (bf16)s0[r];
      myP[row * 32 + 16 + lr] = (bf16)s1[r];
    }
    __syncthreads();
    const v16bf pa = load_a_frag(myP, 32, 0, 0, lane);
    __syncthreads();

    // ---- O += P * V ----
    acc0 = wmma_bf16(pa, load_bt_frag(vh, (size_t)L_, 0,  j0, lane), acc0);
    acc1 = wmma_bf16(pa, load_bt_frag(vh, (size_t)L_, 16, j0, lane), acc1);
    acc2 = wmma_bf16(pa, load_bt_frag(vh, (size_t)L_, 32, j0, lane), acc2);
    acc3 = wmma_bf16(pa, load_bt_frag(vh, (size_t)L_, 48, j0, lane), acc3);

    buf ^= 1;
  }

  // ---- epilogue: normalize and scatter into [B, L, DIM] bf16 ----
  const int hh = bh & 15;
#pragma unroll
  for (int r = 0; r < 8; ++r) {
    const int row = i0 + hf * 8 + r;
    const float inv = 1.0f / l[r];
    bf16* o = attn_out + ((size_t)bb * L_ + row) * DIM_ + hh * HD_;
    o[0  + lr] = (bf16)(acc0[r] * inv);
    o[16 + lr] = (bf16)(acc1[r] * inv);
    o[32 + lr] = (bf16)(acc2[r] * inv);
    o[48 + lr] = (bf16)(acc3[r] * inv);
  }
}

// ---------------------------------------------------------------------------
extern "C" void kernel_launch(void* const* d_in, const int* in_sizes, int n_in,
                              void* d_out, int out_size, void* d_ws, size_t ws_size,
                              hipStream_t stream) {
  const float* x        = (const float*)d_in[0];
  const float* pos_bias = (const float*)d_in[1];
  const int*   mask     = (const int*)d_in[2];
  const float* wq       = (const float*)d_in[3];
  const float* wk       = (const float*)d_in[4];
  const float* wv       = (const float*)d_in[5];
  const float* wo       = (const float*)d_in[6];
  float* out = (float*)d_out;

  char* ws = (char*)d_ws;
  const size_t XB = (size_t)M_ * DIM_ * sizeof(bf16);
  const size_t WT = (size_t)DIM_ * DIM_ * sizeof(bf16);
  bf16* xb  = (bf16*)(ws);
  bf16* wqT = (bf16*)(ws + XB);
  bf16* wkT = (bf16*)(ws + XB + WT);
  bf16* wvT = (bf16*)(ws + XB + 2 * WT);
  bf16* woT = (bf16*)(ws + XB + 3 * WT);
  bf16* qb  = (bf16*)(ws + XB + 4 * WT);
  bf16* kb  = (bf16*)(ws + 2 * XB + 4 * WT);
  bf16* vTb = (bf16*)(ws + 3 * XB + 4 * WT);
  bf16* aob = (bf16*)(ws + 4 * XB + 4 * WT);

  const int nX = M_ * DIM_;
  cvt_f32_bf16<<<(nX + 255) / 256, 256, 0, stream>>>(x, xb, nX);
  transpose_w_bf16<<<DIM_, 256, 0, stream>>>(wq, wqT);
  transpose_w_bf16<<<DIM_, 256, 0, stream>>>(wk, wkT);
  transpose_w_bf16<<<DIM_, 256, 0, stream>>>(wv, wvT);
  transpose_w_bf16<<<DIM_, 256, 0, stream>>>(wo, woT);

  const int gemm_blocks = (M_ / 16) * (DIM_ / 16) / 8;  // 2048
  gemm_wmma_bf16<<<gemm_blocks, 256, 0, stream>>>(xb, wqT, qb,  M_, DIM_, DIM_, 0);
  gemm_wmma_bf16<<<gemm_blocks, 256, 0, stream>>>(xb, wkT, kb,  M_, DIM_, DIM_, 1);
  gemm_wmma_bf16<<<gemm_blocks, 256, 0, stream>>>(xb, wvT, vTb, M_, DIM_, DIM_, 2);

  const int attn_blocks = (B_ * NH_ * (L_ / 16)) / 4;   // 1024 blocks of 128
  const size_t attn_lds = 16384 + 4096;                 // bias dbl-buf + P tiles
  attn_wmma<<<attn_blocks, 128, attn_lds, stream>>>(qb, kb, vTb, pos_bias, mask, aob);

  gemm_wmma_bf16<<<gemm_blocks, 256, 0, stream>>>(aob, woT, out, M_, DIM_, DIM_, 3);
}
